// MambaBottleneckWrapper_3453153706074
// MI455X (gfx1250) — compile-verified, matrix operand order checked by reading, not run on hardware
//
#include <hip/hip_runtime.h>
#include <math.h>

typedef __attribute__((ext_vector_type(16))) __bf16 v16bf;
typedef __attribute__((ext_vector_type(8)))  float  v8f;

#define NB    8
#define LL    1024
#define CC    512
#define DIM   1024
#define NST   16
#define RR    32
#define MTOK  8192   // NB*LL

// ---------- bf16 helpers (bit-level, avoids __hip_bfloat16 ABI differences) ----------
__device__ __forceinline__ unsigned short f2bf(float f) {
  union { float f; unsigned u; } v; v.f = f;
  unsigned u = v.u;
  unsigned lsb = (u >> 16) & 1u;
  u += 0x7fffu + lsb;           // round-to-nearest-even
  return (unsigned short)(u >> 16);
}
__device__ __forceinline__ float bf2f(unsigned short h) {
  union { unsigned u; float f; } v; v.u = ((unsigned)h) << 16;
  return v.f;
}

// gfx1250 async global->LDS copy, 16B per lane, tracked by ASYNCcnt.
// lds_addr: LDS byte address (low 32 bits of a generic __shared__ pointer ==
// LDS aperture offset, ISA sec.10.2). gaddr: 64-bit global address.
__device__ __forceinline__ void async_g2l_b128(unsigned lds_addr, const void* gptr) {
  asm volatile("global_load_async_to_lds_b128 %0, %1, off"
               :: "v"(lds_addr), "v"((unsigned long long)gptr) : "memory");
}
__device__ __forceinline__ void wait_async0() {
  asm volatile("s_wait_asynccnt 0x0" ::: "memory");
}

// ---------- utility kernels ----------
__global__ void k_f32_to_bf16(const float* __restrict__ in, unsigned short* __restrict__ out, int n) {
  int i = blockIdx.x * blockDim.x + threadIdx.x;
  if (i < n) out[i] = f2bf(in[i]);
}

// ---------- LayerNorm over C=512, emit bf16 ----------
__global__ __launch_bounds__(256) void k_ln_in(const float* __restrict__ x,
                                               const float* __restrict__ w,
                                               const float* __restrict__ b,
                                               unsigned short* __restrict__ xn) {
  int tok = blockIdx.x, t = threadIdx.x;
  const float* xp = x + (size_t)tok * CC;
  float v0 = xp[t], v1 = xp[t + 256];
  __shared__ float red[256];
  red[t] = v0 + v1; __syncthreads();
  for (int o = 128; o > 0; o >>= 1) { if (t < o) red[t] += red[t + o]; __syncthreads(); }
  float mu = red[0] * (1.f / 512.f);
  __syncthreads();
  float d0 = v0 - mu, d1 = v1 - mu;
  red[t] = d0 * d0 + d1 * d1; __syncthreads();
  for (int o = 128; o > 0; o >>= 1) { if (t < o) red[t] += red[t + o]; __syncthreads(); }
  float rstd = rsqrtf(red[0] * (1.f / 512.f) + 1e-5f);
  xn[(size_t)tok * CC + t]       = f2bf(d0 * rstd * w[t]       + b[t]);
  xn[(size_t)tok * CC + t + 256] = f2bf(d1 * rstd * w[t + 256] + b[t + 256]);
}

// ---------- generic bf16 WMMA GEMM:  C[M,N] = A[M,K] * W[N,K]^T  ----------
// mode 0: store f32 ; mode 1: softplus(acc + bias[n]) ; mode 2: acc + resid[m,n]
// Requires M%128==0, N%64==0, K%32==0 (always true here).
// LDS tiles stored in WMMA per-lane fragment layout (ISA 7.12.2, 16-bit operands):
//   A 16x32 subtile: lane = m + 16*((k>>3)&1), idx = (k&7) + 8*(k>>4)
//   B 32x16 subtile: same with n in place of m.
// Staging uses GLOBAL_LOAD_ASYNC_TO_LDS_B128 (ASYNCcnt) - no VGPR round-trip.
__global__ __launch_bounds__(256) void k_gemm_bf16(const unsigned short* __restrict__ A,
                                                   const unsigned short* __restrict__ W,
                                                   float* __restrict__ C,
                                                   int M, int N, int K, int mode,
                                                   const float* __restrict__ bias,
                                                   const float* __restrict__ resid) {
  __shared__ __align__(32) unsigned short lA[128 * 32];  // 8 KB
  __shared__ __align__(32) unsigned short lB[64 * 32];   // 4 KB
  const int t    = threadIdx.x;
  const int lane = t & 31;
  const int wv   = t >> 5;        // 0..7
  const int wm   = wv >> 1;       // 0..3 : A subtile pair
  const int wn   = wv & 1;        // 0..1 : B subtile pair
  const int m0   = blockIdx.x * 128;
  const int n0   = blockIdx.y * 64;

  const unsigned ldsA = (unsigned)(unsigned long long)&lA[0];
  const unsigned ldsB = (unsigned)(unsigned long long)&lB[0];

  v8f acc[2][2] = {};

  for (int kt = 0; kt < K; kt += 32) {
    // ---- stage A tile 128x32: 2 async 16B transfers per lane, fragment-swizzled ----
    #pragma unroll
    for (int i = 0; i < 2; ++i) {
      int li  = t + i * 256;          // 0..511
      int row = li >> 2;              // 0..127
      int k0  = (li & 3) << 3;        // 0,8,16,24
      int sub = row >> 4, ml = row & 15;
      int ln2 = ml + (((k0 >> 3) & 1) << 4);
      int idx0 = (k0 >> 4) << 3;
      async_g2l_b128(ldsA + (unsigned)(sub * 512 + ln2 * 16 + idx0) * 2u,
                     A + (size_t)(m0 + row) * K + kt + k0);
    }
    // ---- stage B tile 64x32: 1 async 16B transfer per lane ----
    {
      int row = t >> 2;               // 0..63 (n-local)
      int k0  = (t & 3) << 3;
      int sub = row >> 4, nl = row & 15;
      int ln2 = nl + (((k0 >> 3) & 1) << 4);
      int idx0 = (k0 >> 4) << 3;
      async_g2l_b128(ldsB + (unsigned)(sub * 512 + ln2 * 16 + idx0) * 2u,
                     W + (size_t)(n0 + row) * K + kt + k0);
    }
    // prefetch next K-tile rows (gfx1250 global_prefetch_b8)
    if (kt + 32 < K) {
      __builtin_prefetch(A + (size_t)(m0 + (t & 127)) * K + kt + 32);
      if (t < 64) __builtin_prefetch(W + (size_t)(n0 + t) * K + kt + 32);
    }
    wait_async0();          // s_wait_asynccnt 0: this wave's LDS writes landed
    __syncthreads();        // ...and everyone else's

    v16bf a0 = *(const v16bf*)&lA[(wm * 2 + 0) * 512 + lane * 16];
    v16bf a1 = *(const v16bf*)&lA[(wm * 2 + 1) * 512 + lane * 16];
    v16bf b0 = *(const v16bf*)&lB[(wn * 2 + 0) * 512 + lane * 16];
    v16bf b1 = *(const v16bf*)&lB[(wn * 2 + 1) * 512 + lane * 16];

    acc[0][0] = __builtin_amdgcn_wmma_f32_16x16x32_bf16(false, a0, false, b0, (short)0, acc[0][0], false, false);
    acc[0][1] = __builtin_amdgcn_wmma_f32_16x16x32_bf16(false, a0, false, b1, (short)0, acc[0][1], false, false);
    acc[1][0] = __builtin_amdgcn_wmma_f32_16x16x32_bf16(false, a1, false, b0, (short)0, acc[1][0], false, false);
    acc[1][1] = __builtin_amdgcn_wmma_f32_16x16x32_bf16(false, a1, false, b1, (short)0, acc[1][1], false, false);
    __syncthreads();
  }

  // ---- epilogue (C/D layout: VGPR r -> M = r + 8*(lane>=16), N = lane&15) ----
  const int mhalf = (lane >> 4) << 3;
  const int col0  = lane & 15;
  #pragma unroll
  for (int i = 0; i < 2; ++i)
    #pragma unroll
    for (int j = 0; j < 2; ++j)
      #pragma unroll
      for (int r = 0; r < 8; ++r) {
        int row = m0 + (wm * 2 + i) * 16 + mhalf + r;
        int col = n0 + (wn * 2 + j) * 16 + col0;
        float v = acc[i][j][r];
        if (mode == 1) { v += bias[col]; v = (v > 20.f) ? v : log1pf(__expf(v)); }
        if (mode == 2) v += resid[(size_t)row * N + col];
        C[(size_t)row * N + col] = v;
      }
}

// ---------- depthwise 3x3 conv + bias + SiLU; scatter 4 scan orderings (bf16, token-major) ----------
__global__ __launch_bounds__(256) void k_conv_silu(const float* __restrict__ xz,
                                                   const float* __restrict__ cw,
                                                   const float* __restrict__ cb,
                                                   unsigned short* __restrict__ xs) {
  int gid = blockIdx.x * blockDim.x + threadIdx.x;   // (b,l,d), d fastest
  int d = gid & (DIM - 1);
  int l = (gid >> 10) & (LL - 1);
  int b = gid >> 20;
  int h = l >> 5, w = l & 31;
  float acc = cb[d];
  #pragma unroll
  for (int kh = 0; kh < 3; ++kh) {
    int hh = h + kh - 1;
    if ((unsigned)hh >= 32u) continue;
    #pragma unroll
    for (int kw = 0; kw < 3; ++kw) {
      int ww = w + kw - 1;
      if ((unsigned)ww >= 32u) continue;
      acc += cw[d * 9 + kh * 3 + kw] * xz[((size_t)b * LL + hh * 32 + ww) * 2048 + d];
    }
  }
  float sv = acc / (1.f + __expf(-acc));            // SiLU
  unsigned short bv = f2bf(sv);
  int s1 = (w << 5) | h;                            // transpose ordering
  xs[(((size_t)0  + b) * LL + l)            * DIM + d] = bv;   // k=0: hw
  xs[(((size_t)8  + b) * LL + s1)           * DIM + d] = bv;   // k=1: wh
  xs[(((size_t)16 + b) * LL + (1023 - l))   * DIM + d] = bv;   // k=2: hw reversed
  xs[(((size_t)24 + b) * LL + (1023 - s1))  * DIM + d] = bv;   // k=3: wh reversed
}

// ---------- extract dts part of x_dbl as bf16 for the dt GEMM ----------
__global__ void k_make_dtsin(const float* __restrict__ xdbl, unsigned short* __restrict__ dtsin) {
  int gid = blockIdx.x * blockDim.x + threadIdx.x;   // k*8192*32 + m*32 + r
  int r = gid & 31;
  int m = (gid >> 5) & 8191;
  int k = gid >> 18;
  dtsin[gid] = f2bf(xdbl[((size_t)k * MTOK + m) * 64 + r]);
}

// ---------- selective scan: one block per (k,b), one thread per channel d ----------
// Writes ys in scan order (coalesced bf16 stores); direction unscramble happens
// in k_outnorm via involutive index maps -> no global atomics needed.
__global__ __launch_bounds__(1024) void k_scan(const unsigned short* __restrict__ xs,
                                               const float* __restrict__ dt,
                                               const float* __restrict__ xdbl,
                                               const float* __restrict__ A_logs,
                                               const float* __restrict__ Ds,
                                               unsigned short* __restrict__ ys) {
  int kb = blockIdx.x;            // k*8 + b
  int k = kb >> 3, b = kb & 7;
  int d = threadIdx.x;

  float Arow[NST];
  #pragma unroll
  for (int n = 0; n < NST; ++n) Arow[n] = -__expf(A_logs[((size_t)k * DIM + d) * NST + n]);
  float Dv = Ds[k * DIM + d];
  float h[NST];
  #pragma unroll
  for (int n = 0; n < NST; ++n) h[n] = 0.f;

  __shared__ float sBC[32];       // B_t (16) then C_t (16), shared by all d
  size_t xdbase = ((size_t)k * MTOK + b * LL) * 64;
  size_t tbase  = ((size_t)kb * LL) * DIM + d;

  for (int s = 0; s < LL; ++s) {
    if (threadIdx.x < 32) sBC[threadIdx.x] = xdbl[xdbase + (size_t)s * 64 + 32 + threadIdx.x];
    __syncthreads();
    float dtv = dt[tbase + (size_t)s * DIM];
    float u   = bf2f(xs[tbase + (size_t)s * DIM]);
    float yv  = Dv * u;
    #pragma unroll
    for (int n = 0; n < NST; ++n) {
      float dA = __expf(dtv * Arow[n]);
      h[n] = h[n] * dA + dtv * sBC[n] * u;
      yv  += h[n] * sBC[16 + n];
    }
    ys[tbase + (size_t)s * DIM] = f2bf(yv);
    __syncthreads();
  }
}

// ---------- gather 4 directions + LayerNorm (DI=1024) * SiLU(z), emit bf16 ----------
__global__ __launch_bounds__(256) void k_outnorm(const unsigned short* __restrict__ ys,
                                                 const float* __restrict__ xz,
                                                 const float* __restrict__ w,
                                                 const float* __restrict__ b,
                                                 unsigned short* __restrict__ y2) {
  int tok = blockIdx.x, t = threadIdx.x;
  int bb2 = tok >> 10, p = tok & 1023;
  int pt = ((p & 31) << 5) | (p >> 5);   // spatial transpose (involutive)
  // scan index contributing to spatial p for each direction:
  size_t r0 = (((size_t)0  + bb2) * LL + p)           * DIM;
  size_t r1 = (((size_t)8  + bb2) * LL + pt)          * DIM;
  size_t r2 = (((size_t)16 + bb2) * LL + (1023 - p))  * DIM;
  size_t r3 = (((size_t)24 + bb2) * LL + (1023 - pt)) * DIM;

  float v[4]; float s = 0.f;
  #pragma unroll
  for (int i = 0; i < 4; ++i) {
    int c = t + i * 256;
    v[i] = bf2f(ys[r0 + c]) + bf2f(ys[r1 + c]) + bf2f(ys[r2 + c]) + bf2f(ys[r3 + c]);
    s += v[i];
  }
  __shared__ float red[256];
  red[t] = s; __syncthreads();
  for (int o = 128; o > 0; o >>= 1) { if (t < o) red[t] += red[t + o]; __syncthreads(); }
  float mu = red[0] * (1.f / 1024.f);
  __syncthreads();
  float sq = 0.f;
  #pragma unroll
  for (int i = 0; i < 4; ++i) { float dd = v[i] - mu; sq += dd * dd; }
  red[t] = sq; __syncthreads();
  for (int o = 128; o > 0; o >>= 1) { if (t < o) red[t] += red[t + o]; __syncthreads(); }
  float rstd = rsqrtf(red[0] * (1.f / 1024.f) + 1e-5f);
  #pragma unroll
  for (int i = 0; i < 4; ++i) {
    int c = t + i * 256;
    float z  = xz[(size_t)tok * 2048 + 1024 + c];
    float sz = z / (1.f + __expf(-z));
    y2[(size_t)tok * DIM + c] = f2bf(((v[i] - mu) * rstd * w[c] + b[c]) * sz);
  }
}

// ---------------------------------------------------------------------------
extern "C" void kernel_launch(void* const* d_in, const int* in_sizes, int n_in,
                              void* d_out, int out_size, void* d_ws, size_t ws_size,
                              hipStream_t stream) {
  (void)in_sizes; (void)n_in; (void)out_size; (void)ws_size;
  const float* x          = (const float*)d_in[0];
  const float* norm_w     = (const float*)d_in[1];
  const float* norm_b     = (const float*)d_in[2];
  const float* in_proj_w  = (const float*)d_in[3];
  const float* conv_w     = (const float*)d_in[4];
  const float* conv_b     = (const float*)d_in[5];
  const float* x_proj_w   = (const float*)d_in[6];
  const float* dt_projs_w = (const float*)d_in[7];
  const float* dt_projs_b = (const float*)d_in[8];
  const float* A_logs     = (const float*)d_in[9];
  const float* Ds         = (const float*)d_in[10];
  const float* out_norm_w = (const float*)d_in[11];
  const float* out_norm_b = (const float*)d_in[12];
  const float* out_proj_w = (const float*)d_in[13];
  float* out = (float*)d_out;

  char* wsb = (char*)d_ws;
  size_t off = 0;
  auto take = [&](size_t bytes) -> char* {
    char* p = wsb + off;
    off = (off + bytes + 255) & ~(size_t)255;
    return p;
  };
  unsigned short* xn    = (unsigned short*)take((size_t)MTOK * CC * 2);        //   8 MB
  unsigned short* w_in  = (unsigned short*)take((size_t)2048 * CC * 2);        //   2 MB
  unsigned short* w_xp  = (unsigned short*)take((size_t)4 * 64 * DIM * 2);     // 0.5 MB
  unsigned short* w_dt  = (unsigned short*)take((size_t)4 * DIM * RR * 2);     // .25 MB
  unsigned short* w_out = (unsigned short*)take((size_t)CC * DIM * 2);         //   1 MB
  float*          xz    = (float*)take((size_t)MTOK * 2048 * 4);               //  64 MB
  unsigned short* xs    = (unsigned short*)take((size_t)4 * MTOK * DIM * 2);   //  64 MB
  float*          xdbl  = (float*)take((size_t)4 * MTOK * 64 * 4);             //   8 MB
  unsigned short* dtsin = (unsigned short*)take((size_t)4 * MTOK * RR * 2);    //   2 MB
  float*          dtb   = (float*)take((size_t)4 * MTOK * DIM * 4);            // 128 MB
  unsigned short* ysb   = (unsigned short*)take((size_t)4 * MTOK * DIM * 2);   //  64 MB
  unsigned short* y2    = (unsigned short*)take((size_t)MTOK * DIM * 2);       //  16 MB

  // weights -> bf16
  k_f32_to_bf16<<<(2048 * 512 + 255) / 256, 256, 0, stream>>>(in_proj_w,  w_in,  2048 * 512);
  k_f32_to_bf16<<<(4 * 64 * 1024 + 255) / 256, 256, 0, stream>>>(x_proj_w,  w_xp, 4 * 64 * 1024);
  k_f32_to_bf16<<<(4 * 1024 * 32 + 255) / 256, 256, 0, stream>>>(dt_projs_w, w_dt, 4 * 1024 * 32);
  k_f32_to_bf16<<<(512 * 1024 + 255) / 256, 256, 0, stream>>>(out_proj_w, w_out, 512 * 1024);

  // 1) LayerNorm
  k_ln_in<<<MTOK, 256, 0, stream>>>(x, norm_w, norm_b, xn);
  // 2) in_proj: xz[8192,2048] = xn[8192,512] * in_proj_w^T
  k_gemm_bf16<<<dim3(64, 32), 256, 0, stream>>>(xn, w_in, xz, MTOK, 2048, 512, 0, nullptr, nullptr);
  // 3) depthwise conv + SiLU + 4-direction scatter
  k_conv_silu<<<(NB * LL * DIM) / 256, 256, 0, stream>>>(xz, conv_w, conv_b, xs);
  // 4) x_dbl per direction: [8192,64] = xs_k[8192,1024] * x_proj_w_k^T
  for (int k = 0; k < 4; ++k)
    k_gemm_bf16<<<dim3(64, 1), 256, 0, stream>>>(xs + (size_t)k * MTOK * DIM,
                                                 w_xp + (size_t)k * 64 * DIM,
                                                 xdbl + (size_t)k * MTOK * 64,
                                                 MTOK, 64, 1024, 0, nullptr, nullptr);
  // 5) dt: [8192,1024] = softplus(dts[8192,32] * dt_projs_w_k^T + b_k)
  k_make_dtsin<<<(4 * MTOK * 32) / 256, 256, 0, stream>>>(xdbl, dtsin);
  for (int k = 0; k < 4; ++k)
    k_gemm_bf16<<<dim3(64, 16), 256, 0, stream>>>(dtsin + (size_t)k * MTOK * RR,
                                                  w_dt + (size_t)k * DIM * RR,
                                                  dtb + (size_t)k * MTOK * DIM,
                                                  MTOK, 1024, 32, 1, dt_projs_b + k * DIM, nullptr);
  // 6) selective scan (coalesced direction-ordered ys, no atomics)
  k_scan<<<32, 1024, 0, stream>>>(xs, dtb, xdbl, A_logs, Ds, ysb);
  // 7) gather 4 directions + out-norm * SiLU(z)
  k_outnorm<<<MTOK, 256, 0, stream>>>(ysb, xz, out_norm_w, out_norm_b, y2);
  // 8) out_proj + residual: out[8192,512] = y2[8192,1024] * out_proj_w^T + x
  k_gemm_bf16<<<dim3(64, 8), 256, 0, stream>>>(y2, w_out, out, MTOK, 512, 1024, 2, nullptr, x);
}
